// OptimizedScaleAdaptiveRouter_51840255263055
// MI455X (gfx1250) — matrix operand order repaired and checked by reading, not compiled, not run on hardware
//
#include <hip/hip_runtime.h>

typedef __attribute__((ext_vector_type(16))) _Float16 v16h;
typedef __attribute__((ext_vector_type(8)))  _Float16 v8h;
typedef __attribute__((ext_vector_type(8)))  float    v8f;
typedef __attribute__((ext_vector_type(4)))  float    v4f;

#define TOK_PER_BLK 128
#define DDIM        2048
#define EEXP        64
#define KTILE       32
#define KCHUNK      128
#define NSUB        (KCHUNK / KTILE)     // 4
#define NCHUNK      (DDIM / KCHUNK)      // 16
#define NOISE_STD   0.1f

union AFrag { v16h v; _Float16 h[16]; };
union BFrag { v16h v; v8h q[2]; };
union CFrag { v8f v; float f[8]; };

// Async global->LDS 16B copy (ASYNCcnt path). Per ISA 10.7/15.18, the
// instruction offset is added to BOTH the LDS address (VDST VGPR) and the
// global address (VADDR pair), so one base pair serves 8 issues.
#define ASYNC_CP16(loff, gaddr, OFF)                                         \
    asm volatile("global_load_async_to_lds_b128 %0, %1, off offset:" #OFF    \
                 :: "v"(loff), "v"(gaddr) : "memory")

// Stage a [KCHUNK x EEXP] k-major f16 slice of W into LDS.
// 256 threads: thread -> expert pair (2 rows of W), 16 consecutive k values.
// Packs expert-pair (e, e+1) at the same k into one dword -> ds_store_b32.
__device__ __forceinline__ void stage_w(const float* __restrict__ W,
                                        _Float16* __restrict__ wb,
                                        int k0, int tid)
{
    const int ep = tid >> 3;              // 0..31 expert pair
    const int kg = (tid & 7) * 16;        // k offset within chunk
    const int e  = ep * 2;
    const float* w0 = W + (size_t)e * DDIM + k0 + kg;
    const float* w1 = w0 + DDIM;
    v4f av[4], bv[4];
    #pragma unroll
    for (int q = 0; q < 4; ++q) {
        av[q] = *(const v4f*)(w0 + q * 4);
        bv[q] = *(const v4f*)(w1 + q * 4);
    }
    unsigned* dst = (unsigned*)(wb + (size_t)kg * EEXP + e);
    union H2 { unsigned u; _Float16 h[2]; } p;
    #pragma unroll
    for (int q = 0; q < 4; ++q) {
        #pragma unroll
        for (int j = 0; j < 4; ++j) {
            p.h[0] = (_Float16)av[q][j];
            p.h[1] = (_Float16)bv[q][j];
            dst[(q * 4 + j) * (EEXP / 2)] = p.u;   // stride EEXP f16 per k
        }
    }
}

__global__ __launch_bounds__(256, 2)
void moe_router_wmma_kernel(const float* __restrict__ x,
                            const float* __restrict__ W,
                            const float* __restrict__ scale_condition,
                            const float* __restrict__ noise,
                            const int*   __restrict__ scale_idx_p,
                            float* __restrict__ dispatch,
                            float* __restrict__ probs,
                            int*   __restrict__ selected)
{
    // 32KB. During the K loop: double-buffered f16 W slice [2][KCHUNK][EEXP].
    // After the loop: f32 logits [TOK_PER_BLK][EEXP].
    __shared__ float smem[TOK_PER_BLK * EEXP];
    // 32KB noise slab for this block, filled by async global->LDS copies
    // overlapped with the whole GEMM loop.
    __shared__ float noiselds[TOK_PER_BLK * EEXP];
    _Float16* wbuf = (_Float16*)smem;

    const int tid   = threadIdx.x;
    const int lane  = tid & 31;
    const int wave  = tid >> 5;
    const int lmod  = lane & 15;
    const int lhalf = (lane < 16) ? 0 : 1;

    const float sf = 1.0f + scale_condition[*scale_idx_p] * 0.1f;

    const int rowBlock = blockIdx.x * TOK_PER_BLK;
    const size_t aRowBase = (size_t)(rowBlock + wave * 16 + lmod) * DDIM;

    // ---- kick off async noise staging: 128B per thread (8 x 16B) ----
    {
        const unsigned lo =
            (unsigned)(uintptr_t)&noiselds[(size_t)tid * 32];
        const unsigned long long ga =
            (unsigned long long)(uintptr_t)(noise + (size_t)rowBlock * EEXP
                                                  + (size_t)tid * 32);
        ASYNC_CP16(lo, ga, 0);
        ASYNC_CP16(lo, ga, 16);
        ASYNC_CP16(lo, ga, 32);
        ASYNC_CP16(lo, ga, 48);
        ASYNC_CP16(lo, ga, 64);
        ASYNC_CP16(lo, ga, 80);
        ASYNC_CP16(lo, ga, 96);
        ASYNC_CP16(lo, ga, 112);
    }

    v8f acc[4];
    const v8f vzero = {0.f,0.f,0.f,0.f,0.f,0.f,0.f,0.f};
    acc[0] = vzero; acc[1] = vzero; acc[2] = vzero; acc[3] = vzero;

    // Prologue: stage chunk 0 into buffer 0.
    stage_w(W, wbuf, 0, tid);
    __syncthreads();

    for (int c = 0; c < NCHUNK; ++c) {
        const int cur = c & 1;
        // Overlap: stage next chunk into the other buffer while computing.
        if (c + 1 < NCHUNK)
            stage_w(W, wbuf + (1 - cur) * KCHUNK * EEXP, (c + 1) * KCHUNK, tid);

        const _Float16* wb = wbuf + cur * KCHUNK * EEXP;

        #pragma unroll
        for (int s = 0; s < NSUB; ++s) {
            const int k0 = c * KCHUNK + s * KTILE;

            // ---- A fragment: row = lmod; K per ISA 16-bit A layout
            //      lanes 0-15: K 0-7 & 16-23; lanes 16-31: K 8-15 & 24-31.
            AFrag af;
            {
                const float* ap = x + aRowBase + k0 + lhalf * 8;
                __builtin_prefetch(ap + KCHUNK, 0, 3);   // global_prefetch_b8, WGP scope
                v4f a0 = *(const v4f*)(ap);
                v4f a1 = *(const v4f*)(ap + 4);
                v4f a2 = *(const v4f*)(ap + 16);
                v4f a3 = *(const v4f*)(ap + 20);
                #pragma unroll
                for (int j = 0; j < 4; ++j) {
                    af.h[j]      = (_Float16)(a0[j] * sf);
                    af.h[4 + j]  = (_Float16)(a1[j] * sf);
                    af.h[8 + j]  = (_Float16)(a2[j] * sf);
                    af.h[12 + j] = (_Float16)(a3[j] * sf);
                }
            }

            // ---- B fragments from LDS (lane = N column; 16 f16 of K per lane, same K order as A)
            #pragma unroll
            for (int t = 0; t < 4; ++t) {
                BFrag bf;
                const _Float16* bp =
                    wb + (size_t)(s * KTILE + lhalf * 8) * EEXP + t * 16 + lmod;
                #pragma unroll
                for (int j = 0; j < 8; ++j) {
                    bf.v[j]     = bp[j * EEXP];
                    bf.v[8 + j] = bp[(16 + j) * EEXP];
                }
                acc[t] = __builtin_amdgcn_wmma_f32_16x16x32_f16(
                             false, af.v, false, bf.v, (short)0, acc[t], false, false);
            }
        }
        __syncthreads();   // next-chunk staging complete + this buffer free
    }

    // ---- spill logits: row_local = wave*16 + lhalf*8 + i, n = t*16 + lmod
    #pragma unroll
    for (int t = 0; t < 4; ++t) {
        CFrag cfr; cfr.v = acc[t];
        #pragma unroll
        for (int i = 0; i < 8; ++i) {
            smem[(wave * 16 + lhalf * 8 + i) * EEXP + t * 16 + lmod] = cfr.f[i];
        }
    }
    // Drain this wave's async noise copies, then make everything visible
    // block-wide (each wave waits its own ASYNCcnt; barrier covers the rest).
    asm volatile("s_wait_asynccnt 0x0" ::: "memory");
    __syncthreads();

    // ---- per-token epilogue: one thread per row (threads 0..127)
    if (tid < TOK_PER_BLK) {
        const size_t row = (size_t)rowBlock + tid;
        float l[EEXP];
        #pragma unroll
        for (int q = 0; q < EEXP / 4; ++q) {
            v4f lv = *(const v4f*)(&smem[tid * EEXP + q * 4]);
            v4f nv = *(const v4f*)(&noiselds[tid * EEXP + q * 4]);
            #pragma unroll
            for (int j = 0; j < 4; ++j) l[q * 4 + j] = lv[j] + nv[j] * NOISE_STD;
        }

        float m = l[0];
        #pragma unroll
        for (int e = 1; e < EEXP; ++e) m = fmaxf(m, l[e]);
        float sum = 0.f;
        #pragma unroll
        for (int e = 0; e < EEXP; ++e) { l[e] = __expf(l[e] - m); sum += l[e]; }
        const float inv = 1.f / sum;
        #pragma unroll
        for (int e = 0; e < EEXP; ++e) l[e] *= inv;

        float* pRow = probs + row * EEXP;
        #pragma unroll
        for (int q = 0; q < EEXP / 4; ++q) {
            v4f pv = { l[q*4], l[q*4+1], l[q*4+2], l[q*4+3] };
            *(v4f*)(pRow + q * 4) = pv;
        }

        // top-2 (ties resolve to lower index, matching jax.lax.top_k)
        int i1 = 0, i2 = -1;
        float v1 = l[0], v2 = -1.f;
        #pragma unroll
        for (int e = 1; e < EEXP; ++e) {
            float p = l[e];
            if (p > v1)      { v2 = v1; i2 = i1; v1 = p; i1 = e; }
            else if (p > v2) { v2 = p;  i2 = e; }
        }
        const float wnorm = 1.f / (v1 + v2);
        const float w1 = v1 * wnorm, w2 = v2 * wnorm;

        float* dRow = dispatch + row * EEXP;
        const v4f z4 = {0.f, 0.f, 0.f, 0.f};
        #pragma unroll
        for (int q = 0; q < EEXP / 4; ++q) *(v4f*)(dRow + q * 4) = z4;
        dRow[i1] = w1;
        dRow[i2] = w2;

        selected[row * 2 + 0] = i1;
        selected[row * 2 + 1] = i2;
    }
}

extern "C" void kernel_launch(void* const* d_in, const int* in_sizes, int n_in,
                              void* d_out, int out_size, void* d_ws, size_t ws_size,
                              hipStream_t stream) {
    (void)n_in; (void)out_size; (void)d_ws; (void)ws_size;
    const float* x     = (const float*)d_in[0];
    const float* W     = (const float*)d_in[1];
    const float* sc    = (const float*)d_in[2];
    const float* noise = (const float*)d_in[3];
    const int*   sidx  = (const int*)d_in[4];

    const int rows = in_sizes[3] / EEXP;     // B*S from noise [B,S,E]
    float* out      = (float*)d_out;
    float* dispatch = out;
    float* probsp   = out + (size_t)rows * EEXP;
    int*   selected = (int*)(out + 2 * (size_t)rows * EEXP);

    const int blocks = rows / TOK_PER_BLK;   // 16384/128 = 128
    hipLaunchKernelGGL(moe_router_wmma_kernel, dim3(blocks), dim3(256), 0, stream,
                       x, W, sc, noise, sidx, dispatch, probsp, selected);
}
